// LinearModel_14233521619456
// MI455X (gfx1250) — matrix operand order repaired and checked by reading, not compiled, not run on hardware
//
#include <hip/hip_runtime.h>

// ---------------------------------------------------------------------------
// LinearModel: logits = (gather(We, tokens) + be) @ Wu^T + bu
// M = 4096 (B*T), N = 32000 (V), K = 1024 (E), fp32 in/out.
// Strategy: bf16 hi/lo split (fp32-accurate "bf16x3" GEMM) on
// v_wmma_f32_16x16x32_bf16, with DOUBLE-BUFFERED async global->LDS staging
// (global_load_async_to_lds_b128 + s_wait_asynccnt) of 128x128 block tiles.
// ---------------------------------------------------------------------------

typedef __attribute__((ext_vector_type(16))) __bf16 v16bf;
typedef __attribute__((ext_vector_type(8)))  float  v8f;

#define EMBED  1024
#define VOCAB  32000
#define MTOT   4096          // B*T
#define BK     32            // K-chunk staged per iteration
#define NCHUNK (EMBED / BK)  // 32

typedef unsigned short u16;
typedef unsigned int   u32;

struct __align__(8) US4 { u16 x, y, z, w; };

__device__ __forceinline__ u16 f2bf(float f) {
  u32 u = __builtin_bit_cast(u32, f);
  u += 0x7FFFu + ((u >> 16) & 1u);   // round-to-nearest-even
  return (u16)(u >> 16);
}
__device__ __forceinline__ float bf2f(u16 h) {
  u32 u = ((u32)h) << 16;
  return __builtin_bit_cast(float, u);
}
__device__ __forceinline__ void split2(float x, u16& hi, u16& lo) {
  hi = f2bf(x);
  lo = f2bf(x - bf2f(hi));
}

// low 32 bits of an LDS flat address == workgroup-relative LDS byte offset,
// which is what GLOBAL_LOAD_ASYNC_TO_LDS expects in its VDST operand.
__device__ __forceinline__ u32 lds_off(const void* p) {
  return (u32)(size_t)p;
}

// async 16B copy: global -> LDS, tracked by ASYNCcnt (per-lane address pair)
#define ASYNC_CP16(ldsoff, gp)                                          \
  asm volatile("global_load_async_to_lds_b128 %0, %1, off"              \
               :: "v"(ldsoff), "v"(gp) : "memory")

#if __has_builtin(__builtin_amdgcn_s_wait_asynccnt)
#define WAIT_ASYNC(n) __builtin_amdgcn_s_wait_asynccnt(n)
#else
#define WAIT_ASYNC(n) asm volatile("s_wait_asynccnt %0" :: "n"(n) : "memory")
#endif

// ---------------------------------------------------------------------------
// Prep 1: emb[m][e] = We[e*VOCAB + tok[m]] + be[e]  ->  bf16 hi/lo planes
// ---------------------------------------------------------------------------
__global__ __launch_bounds__(256) void emb_split_kernel(
    const int* __restrict__ tokens, const float* __restrict__ We,
    const float* __restrict__ be, u16* __restrict__ embHi,
    u16* __restrict__ embLo) {
  u32 idx = blockIdx.x * 256u + threadIdx.x;   // 4096*1024 total
  u32 m = idx >> 10, e = idx & 1023u;
  u32 tok = (u32)tokens[m];
  float x = We[(size_t)e * VOCAB + tok] + be[e];
  u16 hi, lo;
  split2(x, hi, lo);
  embHi[idx] = hi;
  embLo[idx] = lo;
}

// ---------------------------------------------------------------------------
// Prep 2: Wu (row-major [V][E]) -> bf16 hi/lo planes, 4 elems/thread
// ---------------------------------------------------------------------------
__global__ __launch_bounds__(256) void wu_split_kernel(
    const float* __restrict__ Wu, u16* __restrict__ WuHi,
    u16* __restrict__ WuLo) {
  u32 i = (blockIdx.x * 256u + threadIdx.x) * 4u;
  float4 x = *(const float4*)&Wu[i];
  US4 hi, lo;
  split2(x.x, hi.x, lo.x);
  split2(x.y, hi.y, lo.y);
  split2(x.z, hi.z, lo.z);
  split2(x.w, hi.w, lo.w);
  *(US4*)&WuHi[i] = hi;
  *(US4*)&WuLo[i] = lo;
}

// ---------------------------------------------------------------------------
// GEMM: C[M][N] = embf32 @ Wu^T + bu via bf16x3 split WMMA.
// Block: 256 threads = 8 wave32, wave grid 4(M) x 2(N).
// Block tile 128x128, wave tile 32x64 (2x4 of 16x16 accum tiles).
// Double-buffered async-to-LDS staging of BK=32 K-chunks.
// ---------------------------------------------------------------------------
union FragBF {
  uint4 q[2];
  v16bf v;
};

__global__ __launch_bounds__(256) void logits_gemm_kernel(
    const u16* __restrict__ embHi, const u16* __restrict__ embLo,
    const u16* __restrict__ WuHi,  const u16* __restrict__ WuLo,
    const float* __restrict__ bu,  float* __restrict__ out) {
  __shared__ __align__(16) u16 sAhi[2][128 * BK];
  __shared__ __align__(16) u16 sAlo[2][128 * BK];
  __shared__ __align__(16) u16 sBhi[2][128 * BK];
  __shared__ __align__(16) u16 sBlo[2][128 * BK];

  const u32 tid    = threadIdx.x;
  const u32 lane   = tid & 31u;
  const u32 waveId = tid >> 5;
  const u32 waveM  = waveId & 3u;   // 0..3  -> 32-row slabs
  const u32 waveN  = waveId >> 2;   // 0..1  -> 64-col slabs
  const u32 half   = lane >> 4;     // 0/1   (lane groups per WMMA layout)
  const u32 lrow   = lane & 15u;

  const u32 mBlock = blockIdx.y * 128u;
  const u32 nBlock = blockIdx.x * 128u;

  v8f acc[2][4];
#pragma unroll
  for (int mt = 0; mt < 2; ++mt)
#pragma unroll
    for (int nt = 0; nt < 4; ++nt) acc[mt][nt] = (v8f){};

  // staging coords: 2 threads per row, 16 bf16 (32B = 2 async b128) each
  const u32 sr = tid >> 1;          // 0..127
  const u32 sh = tid & 1u;          // 0/1
  const size_t gA = (size_t)(mBlock + sr) * EMBED + sh * 16u;
  const size_t gB = (size_t)(nBlock + sr) * EMBED + sh * 16u;
  const u32 sOff = sr * BK + sh * 16u;

  const u16* pAh = embHi + gA;
  const u16* pAl = embLo + gA;
  const u16* pBh = WuHi + gB;
  const u16* pBl = WuLo + gB;

  // ---- issue async staging of one K-chunk into LDS buffer b ----
  auto stage = [&](int k, int b) {
    ASYNC_CP16(lds_off(&sAhi[b][sOff]),     pAh + k);
    ASYNC_CP16(lds_off(&sAhi[b][sOff + 8]), pAh + k + 8);
    ASYNC_CP16(lds_off(&sAlo[b][sOff]),     pAl + k);
    ASYNC_CP16(lds_off(&sAlo[b][sOff + 8]), pAl + k + 8);
    ASYNC_CP16(lds_off(&sBhi[b][sOff]),     pBh + k);
    ASYNC_CP16(lds_off(&sBhi[b][sOff + 8]), pBh + k + 8);
    ASYNC_CP16(lds_off(&sBlo[b][sOff]),     pBl + k);
    ASYNC_CP16(lds_off(&sBlo[b][sOff + 8]), pBl + k + 8);
  };

  stage(0, 0);   // preload chunk 0 into buffer 0

  for (int i = 0; i < NCHUNK; ++i) {
    const int cur = i & 1;
    if (i + 1 < NCHUNK) {
      // prior-iteration tail barrier guarantees nobody still reads buf cur^1
      stage((i + 1) * BK, cur ^ 1);
      WAIT_ASYNC(8);   // 8 newest outstanding = next chunk; current chunk done
    } else {
      WAIT_ASYNC(0);
    }
    __syncthreads();   // current chunk visible to all waves

    // ---- fragment loads (per ISA 16-bit A 16x32 / B 32x16 VGPR layouts) ----
    FragBF aHi[2], aLo[2], bHi[4], bLo[4];
#pragma unroll
    for (int mt = 0; mt < 2; ++mt) {
      const u32 row = waveM * 32u + (u32)mt * 16u + lrow;
      const u32 b0 = row * BK + half * 8u;         // K = half*8 .. +8
      const u32 b1 = row * BK + 16u + half * 8u;   // K = 16+half*8 .. +8
      aHi[mt].q[0] = *(const uint4*)&sAhi[cur][b0];
      aHi[mt].q[1] = *(const uint4*)&sAhi[cur][b1];
      aLo[mt].q[0] = *(const uint4*)&sAlo[cur][b0];
      aLo[mt].q[1] = *(const uint4*)&sAlo[cur][b1];
    }
#pragma unroll
    for (int nt = 0; nt < 4; ++nt) {
      const u32 row = waveN * 64u + (u32)nt * 16u + lrow;
      const u32 b0 = row * BK + half * 8u;         // K = half*8 .. +8, then 16+half*8 .. +8
      bHi[nt].q[0] = *(const uint4*)&sBhi[cur][b0];
      bHi[nt].q[1] = *(const uint4*)&sBhi[cur][b0 + 16];
      bLo[nt].q[0] = *(const uint4*)&sBlo[cur][b0];
      bLo[nt].q[1] = *(const uint4*)&sBlo[cur][b0 + 16];
    }

    // ---- bf16x3 split WMMA: hi*hi + hi*lo + lo*hi ----
#pragma unroll
    for (int mt = 0; mt < 2; ++mt) {
#pragma unroll
      for (int nt = 0; nt < 4; ++nt) {
        acc[mt][nt] = __builtin_amdgcn_wmma_f32_16x16x32_bf16(
            false, aHi[mt].v, false, bHi[nt].v, (short)0, acc[mt][nt], false, false);
        acc[mt][nt] = __builtin_amdgcn_wmma_f32_16x16x32_bf16(
            false, aHi[mt].v, false, bLo[nt].v, (short)0, acc[mt][nt], false, false);
        acc[mt][nt] = __builtin_amdgcn_wmma_f32_16x16x32_bf16(
            false, aLo[mt].v, false, bHi[nt].v, (short)0, acc[mt][nt], false, false);
      }
    }
    __syncthreads();   // all waves done reading buf cur -> re-stageable
  }

  // ---- epilogue: + bu, stream 512MB out (C/D layout: vgpr r -> M=r+8*half) ----
  const u32 m0 = mBlock + waveM * 32u;
  const u32 n0 = nBlock + waveN * 64u;
#pragma unroll
  for (int nt = 0; nt < 4; ++nt) {
    const u32 col = n0 + (u32)nt * 16u + lrow;
    const float bias = bu[col];
#pragma unroll
    for (int mt = 0; mt < 2; ++mt) {
      const u32 rbase = m0 + (u32)mt * 16u + half * 8u;
#pragma unroll
      for (int r = 0; r < 8; ++r) {
        out[(size_t)(rbase + r) * VOCAB + col] = acc[mt][nt][r] + bias;
      }
    }
  }
}

// ---------------------------------------------------------------------------
extern "C" void kernel_launch(void* const* d_in, const int* in_sizes, int n_in,
                              void* d_out, int out_size, void* d_ws, size_t ws_size,
                              hipStream_t stream) {
  const int*   tokens = (const int*)d_in[0];
  const float* We     = (const float*)d_in[1];
  const float* be     = (const float*)d_in[2];
  const float* Wu     = (const float*)d_in[3];
  const float* bu     = (const float*)d_in[4];
  float* out = (float*)d_out;

  // workspace layout (u16 elements)
  u16* ws     = (u16*)d_ws;
  u16* embHi  = ws;                                   //  4096*1024
  u16* embLo  = embHi + (size_t)MTOT * EMBED;         //  4096*1024
  u16* WuHi   = embLo + (size_t)MTOT * EMBED;         // 32000*1024
  u16* WuLo   = WuHi + (size_t)VOCAB * EMBED;         // 32000*1024

  // prep 1: gather + bias + bf16 split of embeddings
  emb_split_kernel<<<(MTOT * EMBED) / 256, 256, 0, stream>>>(
      tokens, We, be, embHi, embLo);

  // prep 2: bf16 split of Wu (once; never in GEMM inner loop)
  wu_split_kernel<<<(VOCAB * EMBED) / (256 * 4), 256, 0, stream>>>(
      Wu, WuHi, WuLo);

  // GEMM: 250 x 32 blocks of 128x128
  dim3 grid(VOCAB / 128, MTOT / 128);
  logits_gemm_kernel<<<grid, 256, 0, stream>>>(embHi, embLo, WuHi, WuLo, bu, out);
}
